// CausualSelfActivation_34797825032625
// MI455X (gfx1250) — matrix operand order repaired and checked by reading, not compiled, not run on hardware
//
#include <hip/hip_runtime.h>

typedef __bf16 bf16;
typedef __attribute__((ext_vector_type(16))) __bf16 v16bf;
typedef __attribute__((ext_vector_type(8)))  __bf16 v8bf;
typedef __attribute__((ext_vector_type(8)))  float  v8f;

#if __has_builtin(__builtin_amdgcn_global_load_async_to_lds_b128)
#define USE_ASYNC_LDS 1
#endif

typedef int v4i __attribute__((vector_size(16)));
typedef __attribute__((address_space(1))) v4i* gvp_t;  // global int4*
typedef __attribute__((address_space(3))) v4i* lvp_t;  // LDS int4*

// 16-byte global -> LDS copy. Async (ASYNCcnt-tracked, VGPR-bypassing) when the
// CDNA5 builtin is available; falls back to load+ds_store otherwise.
static __device__ __forceinline__ void cp_b128(bf16* lds_dst, const bf16* gsrc) {
#if defined(USE_ASYNC_LDS)
  __builtin_amdgcn_global_load_async_to_lds_b128(
      (gvp_t)(uintptr_t)gsrc,
      (lvp_t)(uint32_t)(uintptr_t)lds_dst, 0, 0);
#else
  *(v8bf*)lds_dst = *(const v8bf*)gsrc;
#endif
}

static __device__ __forceinline__ void async_wait0() {
#if defined(USE_ASYNC_LDS)
#if __has_builtin(__builtin_amdgcn_s_wait_asynccnt)
  __builtin_amdgcn_s_wait_asynccnt(0);
#else
  asm volatile("s_wait_asynccnt 0x0" ::: "memory");
#endif
#endif
}

// Load a 16-element bf16 A/B WMMA fragment piece from two 16-byte LDS chunks.
static __device__ __forceinline__ v16bf load_frag(const bf16* p0, const bf16* p1) {
  v8bf a = *(const v8bf*)p0;
  v8bf b = *(const v8bf*)p1;
  return __builtin_shufflevector(a, b, 0,1,2,3,4,5,6,7,8,9,10,11,12,13,14,15);
}

static __device__ __forceinline__ v8f wmma_bf16(v16bf a, v16bf b, v8f c) {
  return __builtin_amdgcn_wmma_f32_16x16x32_bf16(false, a, false, b, (short)0, c, false, false);
}

// ---------------------------------------------------------------------------
// Elementwise fp32 -> bf16
__global__ __launch_bounds__(256) void cvt_f32_bf16(const float* __restrict__ in,
                                                    bf16* __restrict__ out, int n) {
  int i = blockIdx.x * 256 + threadIdx.x;
  if (i < n) out[i] = (bf16)in[i];
}

// out[n*K + k] = (bf16) in[k*N + n]   (transpose + convert, one-time cost)
__global__ __launch_bounds__(256) void transpose_cvt(const float* __restrict__ in,
                                                     bf16* __restrict__ out, int K, int N) {
  int idx = blockIdx.x * 256 + threadIdx.x;
  if (idx < K * N) {
    int k = idx / N, n = idx - k * N;
    out[(size_t)n * K + k] = (bf16)in[idx];
  }
}

// ---------------------------------------------------------------------------
// Tiled bf16 GEMM:  D = A[M,K] * Bt[N,K]^T + bias
// Block tile 128x128, K-tile 32, double-buffered LDS with async global->LDS
// prefetch of tile k+1 overlapping WMMA on tile k.
// 8 waves arranged 2(M) x 4(N); each wave computes 4x2 WMMA 16x16 tiles.
// MODE 0: QKV epilogue (scatter bf16 into Q[bh][t][d], K[bh][t][d], Vt[bh][d][t])
// MODE 1: fp32 output epilogue (out[m*N+n])
#define G_BM 128
#define G_BN 128
#define G_BK 32
#define G_LS 40  // LDS row stride (elements): 32 + 8 pad

template <int MODE>
__global__ __launch_bounds__(256) void gemm_bf16_kernel(
    const bf16* __restrict__ A, const bf16* __restrict__ Bt,
    const float* __restrict__ bias, int M, int N, int K,
    bf16* __restrict__ q_out, bf16* __restrict__ k_out, bf16* __restrict__ vt_out,
    float* __restrict__ out) {
  __shared__ __align__(16) bf16 As[2][G_BM * G_LS];
  __shared__ __align__(16) bf16 Bs[2][G_BN * G_LS];

  const int tid  = threadIdx.x;
  const int lane = tid & 31;
  const int wave = tid >> 5;
  const int waveM = wave >> 2;   // 0..1 -> 64 rows each
  const int waveN = wave & 3;    // 0..3 -> 32 cols each
  const int l15  = lane & 15;
  const int lh   = lane >> 4;

  const int bm = blockIdx.y * G_BM;
  const int bn = blockIdx.x * G_BN;

  v8f acc[4][2];
  for (int i = 0; i < 4; i++)
    for (int j = 0; j < 2; j++) acc[i][j] = (v8f)0.0f;

  const int ldr = tid >> 1;         // 0..127
  const int ldc = (tid & 1) * 16;   // 0 or 16

  auto stage = [&](int buf, int k0) {
    const bf16* ag = A  + (size_t)(bm + ldr) * K + k0 + ldc;
    const bf16* bg = Bt + (size_t)(bn + ldr) * K + k0 + ldc;
    cp_b128(&As[buf][ldr * G_LS + ldc],     ag);
    cp_b128(&As[buf][ldr * G_LS + ldc + 8], ag + 8);
    cp_b128(&Bs[buf][ldr * G_LS + ldc],     bg);
    cp_b128(&Bs[buf][ldr * G_LS + ldc + 8], bg + 8);
  };

  const int nk = K / G_BK;
  stage(0, 0);
  async_wait0();
  __syncthreads();

  for (int kt = 0; kt < nk; kt++) {
    if (kt + 1 < nk) stage((kt + 1) & 1, (kt + 1) * G_BK);

    const bf16* Ab = As[kt & 1];
    const bf16* Bb = Bs[kt & 1];

    v16bf af[4];
#pragma unroll
    for (int mt = 0; mt < 4; mt++) {
      const bf16* base = &Ab[(waveM * 64 + mt * 16 + l15) * G_LS];
      af[mt] = load_frag(base + lh * 8, base + 16 + lh * 8);
    }
    v16bf bfr[2];
#pragma unroll
    for (int nt = 0; nt < 2; nt++) {
      const bf16* base = &Bb[(waveN * 32 + nt * 16 + l15) * G_LS];
      bfr[nt] = load_frag(base + lh * 8, base + 16 + lh * 8);
    }
#pragma unroll
    for (int mt = 0; mt < 4; mt++)
#pragma unroll
      for (int nt = 0; nt < 2; nt++)
        acc[mt][nt] = wmma_bf16(af[mt], bfr[nt], acc[mt][nt]);

    async_wait0();
    __syncthreads();
  }

  // Epilogue. C/D layout: element r of acc holds (m = r + 8*lh, n = l15).
#pragma unroll
  for (int mt = 0; mt < 4; mt++) {
#pragma unroll
    for (int nt = 0; nt < 2; nt++) {
      const int n  = bn + waveN * 32 + nt * 16 + l15;
      const float bv = bias[n];
#pragma unroll
      for (int r = 0; r < 8; r++) {
        const int m = bm + waveM * 64 + mt * 16 + r + 8 * lh;
        const float v = acc[mt][nt][r] + bv;
        if (MODE == 0) {
          const int which = n >> 10;   // 0=q 1=k 2=v
          const int c = n & 1023;
          const int h = c >> 6;
          const int d = c & 63;
          const int b = m >> 11;       // m / 2048
          const int t = m & 2047;
          const int bh = b * 16 + h;
          if (which == 0)      q_out[((size_t)bh * 2048 + t) * 64 + d] = (bf16)v;
          else if (which == 1) k_out[((size_t)bh * 2048 + t) * 64 + d] = (bf16)v;
          else                 vt_out[((size_t)bh * 64 + d) * 2048 + t] = (bf16)v;
        } else {
          out[(size_t)m * N + n] = v;
        }
      }
    }
  }
}

// ---------------------------------------------------------------------------
// Flash attention (causal). Grid: (T/128, B*H). Block: 256 threads = 8 waves.
// Each wave owns 16 q rows. KV streamed in 64-wide blocks through double-
// buffered LDS (async prefetch of block j+1 overlaps WMMA on block j). The Q
// staging region is reused as the second KV buffer once Q fragments are in
// registers, keeping total LDS at 54 KB.
#define FA_BM 128
#define FA_BN 64
#define FA_HD 64
#define FA_ST 72  // LDS row stride
#define FA_QSZ (FA_BM * FA_ST)   // 9216 elements (also size of one KV set)
#define FA_KSZ (FA_BN * FA_ST)   // 4608 elements

__global__ __launch_bounds__(256) void flash_attn_kernel(
    const bf16* __restrict__ Q,   // [BH][T][64]
    const bf16* __restrict__ Kk,  // [BH][T][64]
    const bf16* __restrict__ Vt,  // [BH][64][T]
    bf16* __restrict__ Y,         // [B][T][1024]
    int T) {
  __shared__ __align__(16) bf16 smem[3 * FA_QSZ];  // [Ps | KV0 | KV1(=Qs)]
  bf16* const Ps = smem;
  bf16* const kv0 = smem + FA_QSZ;
  bf16* const kv1 = smem + 2 * FA_QSZ;
  bf16* const Qs  = kv1;  // Q staged here, region recycled as KV buffer 1

  const int tid  = threadIdx.x;
  const int lane = tid & 31;
  const int wave = tid >> 5;   // wave owns q rows [wave*16, wave*16+16)
  const int l15  = lane & 15;
  const int lh   = lane >> 4;

  const int bh = blockIdx.y;           // 0..31
  const int q0 = blockIdx.x * FA_BM;
  const int b  = bh >> 4;
  const int h  = bh & 15;

  auto stage_kv = [&](bf16* kvb, int kv0i) {
    const int r = tid >> 2;          // 0..63
    const int c = (tid & 3) * 16;    // 0,16,32,48
    bf16* Kb = kvb;
    bf16* Vb = kvb + FA_KSZ;
    const bf16* kg = Kk + ((size_t)bh * T + kv0i + r) * FA_HD + c;
    cp_b128(&Kb[r * FA_ST + c],     kg);
    cp_b128(&Kb[r * FA_ST + c + 8], kg + 8);
    const bf16* vg = Vt + ((size_t)bh * FA_HD + r) * T + kv0i + c;
    cp_b128(&Vb[r * FA_ST + c],     vg);
    cp_b128(&Vb[r * FA_ST + c + 8], vg + 8);
  };

  // Stage Q block and first KV block.
  {
    const int r = tid >> 1;
    const int c = (tid & 1) * 32;
    const bf16* g = Q + ((size_t)bh * T + q0 + r) * FA_HD + c;
#pragma unroll
    for (int i = 0; i < 32; i += 8)
      cp_b128(&Qs[r * FA_ST + c + i], g + i);
  }
  stage_kv(kv0, 0);
  async_wait0();
  __syncthreads();

  // Preload this wave's Q A-fragments (d = 0..31, 32..63).
  v16bf qf[2];
  {
    const bf16* base = &Qs[(wave * 16 + l15) * FA_ST];
#pragma unroll
    for (int ks = 0; ks < 2; ks++)
      qf[ks] = load_frag(base + ks * 32 + lh * 8, base + ks * 32 + 16 + lh * 8);
  }
  __syncthreads();  // all Q reads done before Qs region is recycled as KV1

  v8f o[4];
  for (int i = 0; i < 4; i++) o[i] = (v8f)0.0f;
  float mrow[8], lrow[8];
#pragma unroll
  for (int r = 0; r < 8; r++) { mrow[r] = -1e30f; lrow[r] = 0.0f; }

  const float scale = 0.125f;  // 1/sqrt(64)
  const int jmax = (q0 + FA_BM - 1) / FA_BN;

  for (int j = 0; j <= jmax; j++) {
    const int kvbase = j * FA_BN;
    if (j + 1 <= jmax) stage_kv((j & 1) ? kv0 : kv1, (j + 1) * FA_BN);

    const bf16* Ks = (j & 1) ? kv1 : kv0;
    const bf16* Vs = Ks + FA_KSZ;

    // S = Q * K^T  (wave: 16 q rows x 64 kv cols = 4 N-tiles, 2 K-steps)
    v8f s[4];
    for (int nt = 0; nt < 4; nt++) s[nt] = (v8f)0.0f;
#pragma unroll
    for (int ks = 0; ks < 2; ks++) {
#pragma unroll
      for (int nt = 0; nt < 4; nt++) {
        const bf16* base = &Ks[(nt * 16 + l15) * FA_ST];
        v16bf kf = load_frag(base + ks * 32 + lh * 8, base + ks * 32 + 16 + lh * 8);
        s[nt] = wmma_bf16(qf[ks], kf, s[nt]);
      }
    }

    // Causal mask, scale, online softmax; stash P into LDS as bf16.
#pragma unroll
    for (int r = 0; r < 8; r++) {
      const int q = q0 + wave * 16 + r + 8 * lh;
      float sv[4];
      float rmax = -1e30f;
#pragma unroll
      for (int nt = 0; nt < 4; nt++) {
        const int k = kvbase + nt * 16 + l15;
        float v = s[nt][r] * scale;
        if (k > q) v = -1e30f;
        sv[nt] = v;
        rmax = fmaxf(rmax, v);
      }
#pragma unroll
      for (int off = 1; off < 16; off <<= 1)
        rmax = fmaxf(rmax, __shfl_xor(rmax, off, 32));
      const float mnew  = fmaxf(mrow[r], rmax);
      const float alpha = __expf(mrow[r] - mnew);
      mrow[r] = mnew;
      float rsum = 0.0f;
#pragma unroll
      for (int nt = 0; nt < 4; nt++) {
        const float p = __expf(sv[nt] - mnew);
        sv[nt] = p;
        rsum += p;
      }
#pragma unroll
      for (int off = 1; off < 16; off <<= 1)
        rsum += __shfl_xor(rsum, off, 32);
      lrow[r] = lrow[r] * alpha + rsum;
#pragma unroll
      for (int nt = 0; nt < 4; nt++) o[nt][r] *= alpha;
#pragma unroll
      for (int nt = 0; nt < 4; nt++)
        Ps[(wave * 16 + r + 8 * lh) * FA_ST + nt * 16 + l15] = (bf16)sv[nt];
    }
    __syncthreads();

    // O += P * V   (A-frag from Ps, B-frag from Vs)
#pragma unroll
    for (int ks = 0; ks < 2; ks++) {
      const bf16* pb = &Ps[(wave * 16 + l15) * FA_ST];
      v16bf pf = load_frag(pb + ks * 32 + lh * 8, pb + ks * 32 + 16 + lh * 8);
#pragma unroll
      for (int nt = 0; nt < 4; nt++) {
        const bf16* vb = &Vs[(nt * 16 + l15) * FA_ST];
        v16bf vf = load_frag(vb + ks * 32 + lh * 8, vb + ks * 32 + 16 + lh * 8);
        o[nt] = wmma_bf16(pf, vf, o[nt]);
      }
    }
    async_wait0();
    __syncthreads();
  }

  // Normalize and write merged-head bf16 output row-slice.
#pragma unroll
  for (int r = 0; r < 8; r++) {
    const float inv = 1.0f / lrow[r];
    const int q = q0 + wave * 16 + r + 8 * lh;
    const size_t row = ((size_t)b * T + q) * 1024 + (size_t)h * 64;
#pragma unroll
    for (int nt = 0; nt < 4; nt++)
      Y[row + nt * 16 + l15] = (bf16)(o[nt][r] * inv);
  }
}

// ---------------------------------------------------------------------------
extern "C" void kernel_launch(void* const* d_in, const int* in_sizes, int n_in,
                              void* d_out, int out_size, void* d_ws, size_t ws_size,
                              hipStream_t stream) {
  const float* x      = (const float*)d_in[0];
  const float* W_attn = (const float*)d_in[1];
  const float* b_attn = (const float*)d_in[2];
  const float* W_proj = (const float*)d_in[3];
  const float* b_proj = (const float*)d_in[4];
  float* out = (float*)d_out;

  const int B = 2, T = 2048, C = 1024, H = 16;
  const int M = B * T;  // 4096

  char* ws = (char*)d_ws;
  bf16* Xbf = (bf16*)ws;  ws += (size_t)M * C * 2;        // 8 MB
  bf16* WTa = (bf16*)ws;  ws += (size_t)3 * C * C * 2;    // 6 MB
  bf16* WTp = (bf16*)ws;  ws += (size_t)C * C * 2;        // 2 MB
  bf16* Qb  = (bf16*)ws;  ws += (size_t)M * C * 2;        // 8 MB
  bf16* Kb  = (bf16*)ws;  ws += (size_t)M * C * 2;        // 8 MB
  bf16* Vtb = (bf16*)ws;  ws += (size_t)M * C * 2;        // 8 MB
  bf16* Yb  = (bf16*)ws;  ws += (size_t)M * C * 2;        // 8 MB

  cvt_f32_bf16<<<(M * C + 255) / 256, 256, 0, stream>>>(x, Xbf, M * C);
  transpose_cvt<<<(C * 3 * C + 255) / 256, 256, 0, stream>>>(W_attn, WTa, C, 3 * C);
  transpose_cvt<<<(C * C + 255) / 256, 256, 0, stream>>>(W_proj, WTp, C, C);

  dim3 g1(3 * C / G_BN, M / G_BM);
  gemm_bf16_kernel<0><<<g1, 256, 0, stream>>>(Xbf, WTa, b_attn, M, 3 * C, C,
                                              Qb, Kb, Vtb, nullptr);

  dim3 g2(T / FA_BM, B * H);
  flash_attn_kernel<<<g2, 256, 0, stream>>>(Qb, Kb, Vtb, Yb, T);

  dim3 g3(C / G_BN, M / G_BM);
  gemm_bf16_kernel<1><<<g3, 256, 0, stream>>>(Yb, WTp, b_proj, M, C, C,
                                              nullptr, nullptr, nullptr, out);
}
